// ASBIGCN_66494683676825
// MI455X (gfx1250) — matrix operand order repaired and checked by reading, not compiled, not run on hardware
//
#include <hip/hip_runtime.h>
#include <hip/hip_bf16.h>

#define BB 32
#define SS 256
#define NN 160
#define PP 4
#define HH 384
#define DD 768
#define POLN 3
#define H4 1536
#define H2 768
#define LGLD 192   // padded leading dim for attention logits (multiple of 64)

typedef _Float16 f16;
typedef __attribute__((ext_vector_type(16))) _Float16 v16h;
typedef __attribute__((ext_vector_type(8)))  float    v8f;

#define WMMA16(a,b,c) __builtin_amdgcn_wmma_f32_16x16x32_f16(false,(a),false,(b),(short)0,(c),false,false)

__device__ __forceinline__ float sigf(float x){ return 1.f/(1.f+expf(-x)); }

// ---- WMMA fragment loaders (CDNA5 16-bit layouts, ISA 7.12.2) ----
// A: 16x32 (MxK), row-major memory, lda in elements.
__device__ __forceinline__ v16h load_a_frag(const f16* __restrict__ A, int lda){
  int lane = threadIdx.x & 31;
  int m = lane & 15;
  int kb = (lane >> 4) << 3;               // 0 or 8
  const f16* p0 = A + (long)m*lda + kb;    // K = kb .. kb+7
  const f16* p1 = p0 + 16;                 // K = kb+16 .. kb+23
  v16h r;
#pragma unroll
  for (int i=0;i<8;i++){ r[i]=p0[i]; r[8+i]=p1[i]; }
  return r;
}
// B: 32x16 (KxN) from a row-major N x K operand (per-lane K contiguous).
__device__ __forceinline__ v16h load_b_frag_nk(const f16* __restrict__ W, int ldw){
  int lane = threadIdx.x & 31;
  int n = lane & 15;
  int kb = (lane >> 4) << 3;               // 0 or 8
  const f16* p = W + (long)n*ldw + kb;
  v16h r;
#pragma unroll
  for (int i=0;i<8;i++){ r[i]=p[i]; r[8+i]=p[16+i]; }
  return r;
}

// ---- batched GEMM: C[MxN] = A[MxK] * W[NxK]^T (+bias) ----
// block = 64 threads (2 waves: M rows 0-15 / 16-31). Each wave owns NT*16 cols.
// REQUIRES: M % 32 == 0, N % (16*NT) == 0, K % 32 == 0. Branch-free inner loop.
template<int NT>
__global__ void gemm_a_bt_t(const f16* __restrict__ A, const f16* __restrict__ W,
                            const float* __restrict__ bias, float* __restrict__ C,
                            int M, int N, int K, long sA, long sW, long sC){
  int z    = blockIdx.z;
  int wave = threadIdx.x >> 5;
  int lane = threadIdx.x & 31;
  int m0 = blockIdx.y*32 + wave*16;
  int n0 = blockIdx.x*(16*NT);
  const f16* Ab = A + z*sA + (long)m0*K;
  const f16* Wb = W + z*sW + (long)n0*K;
  float*     Cb = C + z*sC;
  int nc = lane & 15, mo = (lane>>4)*8;
  v8f acc[NT];
#pragma unroll
  for (int i=0;i<NT;i++){
    float bv = bias ? bias[n0 + i*16 + nc] : 0.f;
#pragma unroll
    for (int r=0;r<8;r++) acc[i][r] = bv;
  }
  for (int k0=0;k0<K;k0+=32){
    v16h a = load_a_frag(Ab + k0, K);
    if (k0 + 32 < K) __builtin_prefetch(Ab + k0 + 32, 0, 0);
#pragma unroll
    for (int i=0;i<NT;i++){
      v16h b = load_b_frag_nk(Wb + (long)i*16*K + k0, K);
      acc[i] = WMMA16(a, b, acc[i]);
    }
  }
#pragma unroll
  for (int i=0;i<NT;i++){
#pragma unroll
    for (int r=0;r<8;r++) Cb[(long)(m0+mo+r)*N + n0 + i*16 + nc] = acc[i][r];
  }
}

// ---- tiled transpose with down-convert: per batch, src[R x C] f32 -> dst[C x R] f16 ----
// grid (C/32, R/32, batches), block 256 (32x8). Coalesced via LDS tile.
__global__ void transpose_f32_f16(const float* __restrict__ src, f16* __restrict__ dst,
                                  int R, int C){
  __shared__ float tile[32][33];
  int z = blockIdx.z;
  const float* S = src + (long)z*R*C;
  f16*         D = dst + (long)z*R*C;
  int c0 = blockIdx.x*32, r0 = blockIdx.y*32;
  int tx = threadIdx.x & 31, ty = threadIdx.x >> 5;   // 32 x 8
#pragma unroll
  for (int j=0;j<32;j+=8)
    tile[ty+j][tx] = S[(long)(r0+ty+j)*C + c0 + tx];
  __syncthreads();
#pragma unroll
  for (int j=0;j<32;j+=8)
    D[(long)(c0+ty+j)*R + r0 + tx] = (f16)tile[tx][ty+j];
}

// ---- LSTM gate GEMM: g[32 x 1536] = xg[b, t(b), :] + h[32x384] * Whh[1536x384]^T ----
// grid.x = 1536/64 = 24, block = 64 threads (2 waves, rows 0-15 / 16-31).
// The 64x384 f16 Whh panel (48KB, contiguous) is staged into LDS with
// GLOBAL_LOAD_ASYNC_TO_LDS_B128 (ASYNCcnt), then B-frags come from LDS.
__global__ void lstm_gate_gemm(const f16* __restrict__ h, const f16* __restrict__ Whh,
                               const float* __restrict__ xg, const int* __restrict__ lens,
                               float* __restrict__ g, int s, int rev, int T){
  __shared__ f16 wsh[64*HH];                       // 48 KB panel
  int wave = threadIdx.x >> 5;
  int lane = threadIdx.x & 31;
  int n0 = blockIdx.x*64;
  int mbase = wave*16;
  int nc = lane & 15, mo = (lane>>4)*8;

  // async copy: 64*384*2 = 49152 bytes; 64 threads x 48 x 16B
  {
    unsigned ldsbase = (unsigned)(size_t)(&wsh[0]);
    unsigned long gbase = (unsigned long)(const void*)(Whh + (long)n0*HH);
#pragma unroll 4
    for (int i=0;i<48;i++){
      unsigned      la = ldsbase + threadIdx.x*16u + (unsigned)i*1024u;
      unsigned long ga = gbase   + threadIdx.x*16u + (unsigned)i*1024u;
      asm volatile("global_load_async_to_lds_b128 %0, %1, off"
                   :: "v"(la), "v"(ga) : "memory");
    }
  }

  // accumulator init from xg while the async copy is in flight
  v8f acc[4];
#pragma unroll
  for (int i=0;i<4;i++){
#pragma unroll
    for (int r=0;r<8;r++){
      int b = mbase + mo + r;
      int len = lens[b];
      int t = rev ? (s < len ? len-1-s : s) : s;
      acc[i][r] = xg[((long)b*T + t)*H4 + n0 + i*16 + nc];
    }
  }

  asm volatile("s_wait_asynccnt 0x0" ::: "memory");
  __syncthreads();

#pragma unroll
  for (int k0=0;k0<HH;k0+=32){
    v16h a = load_a_frag(h + (long)mbase*HH + k0, HH);
#pragma unroll
    for (int i=0;i<4;i++){
      v16h b = load_b_frag_nk(&wsh[(long)i*16*HH + k0], HH);
      acc[i] = WMMA16(a, b, acc[i]);
    }
  }
#pragma unroll
  for (int i=0;i<4;i++)
#pragma unroll
    for (int r=0;r<8;r++)
      g[(long)(mbase+mo+r)*H4 + n0 + i*16 + nc] = acc[i][r];
}

// ---- LSTM elementwise update ----
__global__ void lstm_update(const float* __restrict__ g, float* __restrict__ h,
                            float* __restrict__ c, f16* __restrict__ hbf,
                            float* __restrict__ out, const int* __restrict__ lens,
                            int s, int rev, int T, int ldout, int colofs){
  int idx = blockIdx.x*blockDim.x + threadIdx.x;
  if (idx >= BB*HH) return;
  int b = idx / HH, j = idx % HH;
  const float* gb = g + (long)b*H4;
  float ig = sigf(gb[j]);
  float fg = sigf(gb[HH + j]);
  float gg = tanhf(gb[2*HH + j]);
  float og = sigf(gb[3*HH + j]);
  float cn = fg*c[idx] + ig*gg;
  float hn = og*tanhf(cn);
  int len = lens[b];
  float m = (s < len) ? 1.f : 0.f;
  float hh = m*hn + (1.f-m)*h[idx];
  float cc = m*cn + (1.f-m)*c[idx];
  h[idx]=hh; c[idx]=cc; hbf[idx]=(f16)hh;
  int t = rev ? (s < len ? len-1-s : s) : s;
  out[((long)b*T + t)*ldout + colofs + j] = hh*m;
}

__global__ void f32_to_f16_kernel(const float* __restrict__ s, f16* __restrict__ d, long n){
  long i = (long)blockIdx.x*blockDim.x + threadIdx.x;
  if (i < n) d[i] = (f16)s[i];
}

__global__ void embed_kernel(const int* __restrict__ tok, const float* __restrict__ emb,
                             f16* __restrict__ x0){
  int r = blockIdx.x;
  int t = tok[r];
  const float* e = emb + (long)t*DD;
  f16* o = x0 + (long)r*DD;
  for (int d=threadIdx.x; d<DD; d+=blockDim.x) o[d] = (f16)e[d];
}

__global__ void textlen_kernel(const int* __restrict__ tok, int* __restrict__ len){
  __shared__ int sh[256];
  int b = blockIdx.x;
  int cnt = 0;
  for (int t=threadIdx.x; t<SS; t+=blockDim.x) cnt += (tok[b*SS+t] != 0) ? 1 : 0;
  sh[threadIdx.x]=cnt; __syncthreads();
  for (int o=128;o>0;o>>=1){ if (threadIdx.x<o) sh[threadIdx.x]+=sh[threadIdx.x+o]; __syncthreads(); }
  if (threadIdx.x==0) len[b]=sh[0];
}

__global__ void cumsum_kernel(const float* __restrict__ o1, float* __restrict__ cs){
  int idx = blockIdx.x*blockDim.x + threadIdx.x;
  if (idx >= BB*H2) return;
  int b = idx / H2, d = idx % H2;
  const float* src = o1 + (long)b*SS*H2 + d;
  float* dst = cs + (long)b*SS*H2 + d;
  dst[0] = 0.f;
  float run = 0.f;
  for (int t=1;t<SS;t++){ run += src[(long)t*H2]; dst[(long)t*H2] = run; }
}

__global__ void tmps_kernel(const float* __restrict__ cs, const int* __restrict__ ts,
                            const int* __restrict__ te, const int* __restrict__ tl,
                            float* __restrict__ tf, f16* __restrict__ th){
  long idx = (long)blockIdx.x*blockDim.x + threadIdx.x;
  if (idx >= (long)BB*NN*H2) return;
  int d = (int)(idx % H2);
  int i = (int)((idx / H2) % NN);
  int b = (int)(idx / ((long)NN*H2));
  int a = ts[b*NN+i], e = te[b*NN+i];
  const float* c = cs + (long)b*SS*H2 + d;
  float v = c[(long)e*H2] - c[(long)a*H2];
  if (i >= tl[b]) v = 0.f;
  tf[idx]=v; th[idx]=(f16)v;
}

__global__ void adj_kernel(const float* __restrict__ a1, const float* __restrict__ a2,
                           float* __restrict__ af, f16* __restrict__ ah){
  long idx = (long)blockIdx.x*blockDim.x + threadIdx.x;
  if (idx >= (long)BB*NN*NN) return;
  float v = a1[idx] + a2[idx];
  v = (v >= 1.f) ? 1.f : v;
  af[idx]=v; ah[idx]=(f16)v;
}

__global__ void denom_kernel(const float* __restrict__ af, float* __restrict__ den){
  __shared__ float sh[256];
  long row = blockIdx.x;
  float s = 0.f;
  for (int m=threadIdx.x; m<NN; m+=blockDim.x) s += af[row*NN+m];
  sh[threadIdx.x]=s; __syncthreads();
  for (int o=128;o>0;o>>=1){ if (threadIdx.x<o) sh[threadIdx.x]+=sh[threadIdx.x+o]; __syncthreads(); }
  if (threadIdx.x==0) den[row]=sh[0]+1.f;
}

// relu(tmp2/denom)+b_gc -> LayerNorm -> + residual (in-place on out)
__global__ void gcn_post_kernel(const float* __restrict__ t2, const float* __restrict__ den,
                                const float* __restrict__ bgc, const float* __restrict__ lg,
                                const float* __restrict__ lb, float* __restrict__ out){
  __shared__ float sh[256];
  long row = blockIdx.x;
  float dv = den[row];
  const float* r = t2 + row*H2;
  float* o = out + row*H2;
  float v[3]; float s=0.f;
#pragma unroll
  for (int j=0;j<3;j++){
    int d = j*256 + threadIdx.x;
    float x = r[d]/dv; x = x > 0.f ? x : 0.f; x += bgc[d];
    v[j]=x; s+=x;
  }
  sh[threadIdx.x]=s; __syncthreads();
  for (int ofs=128;ofs>0;ofs>>=1){ if (threadIdx.x<ofs) sh[threadIdx.x]+=sh[threadIdx.x+ofs]; __syncthreads(); }
  float mu = sh[0]/(float)H2; __syncthreads();
  s = 0.f;
#pragma unroll
  for (int j=0;j<3;j++){ float d0=v[j]-mu; s+=d0*d0; }
  sh[threadIdx.x]=s; __syncthreads();
  for (int ofs=128;ofs>0;ofs>>=1){ if (threadIdx.x<ofs) sh[threadIdx.x]+=sh[threadIdx.x+ofs]; __syncthreads(); }
  float rstd = rsqrtf(sh[0]/(float)H2 + 1e-5f);
#pragma unroll
  for (int j=0;j<3;j++){
    int d = j*256 + threadIdx.x;
    o[d] = (v[j]-mu)*rstd*lg[d] + lb[d] + o[d];
  }
}

// logits stored with padded leading dim LGLD; valid cols [0, NN)
__global__ void softmax_kernel(const float* __restrict__ lgt, const int* __restrict__ tl,
                               f16* __restrict__ att){
  __shared__ float sh[256];
  long row = blockIdx.x;
  int b = blockIdx.x / NN;
  int m = threadIdx.x;
  float x = -3.0e38f;
  if (m < NN){
    x = lgt[row*LGLD + m];
    if (m >= tl[b]) x += -1e20f;
  }
  sh[m]=x; __syncthreads();
  for (int o=128;o>0;o>>=1){ if (m<o) sh[m]=fmaxf(sh[m],sh[m+o]); __syncthreads(); }
  float mx = sh[0]; __syncthreads();
  float e = (m<NN) ? expf(x-mx) : 0.f;
  sh[m]=e; __syncthreads();
  for (int o=128;o>0;o>>=1){ if (m<o) sh[m]+=sh[m+o]; __syncthreads(); }
  float sum = sh[0];
  if (m<NN) att[row*NN+m] = (f16)(e/sum);
}

__global__ void text_kernel(const float* __restrict__ al, const float* __restrict__ ba,
                            const float* __restrict__ gout, const int* __restrict__ tl,
                            float* __restrict__ text){
  long idx = (long)blockIdx.x*blockDim.x + threadIdx.x;
  if (idx >= (long)BB*NN*H2) return;
  int d = (int)(idx % H2);
  long row = idx / H2;
  int n = (int)(row % NN);
  int b = (int)(row / NN);
  float msk = (n < tl[b]) ? 1.f : 0.f;
  text[row*(2*H2) + d]      = al[idx]*msk + ba[d];
  text[row*(2*H2) + H2 + d] = gout[idx];
}

__global__ void span_feat_kernel(const float* __restrict__ text, const float* __restrict__ xseq,
                                 const float* __restrict__ hf, const float* __restrict__ hb,
                                 const int* __restrict__ sps, const int* __restrict__ spe,
                                 float* __restrict__ feat){
  int b = blockIdx.x;
  int s0 = sps[b*PP], s1 = spe[b*PP];
  float* fb = feat + (long)b*(4*H2);
  for (int d=threadIdx.x; d<H2; d+=blockDim.x)
    fb[d] = (d < HH) ? hf[b*HH+d] : hb[b*HH+d-HH];
  for (int d=threadIdx.x; d<2*H2; d+=blockDim.x){
    float mx = -3.0e38f;
    for (int n=s0;n<s1;n++) mx = fmaxf(mx, text[((long)b*NN+n)*(2*H2) + d]);
    fb[H2 + d] = mx;
  }
  for (int d=threadIdx.x; d<H2; d+=blockDim.x){
    float sm = 0.f;
    for (int n=s0;n<s1;n++) sm += xseq[((long)b*NN+n)*H2 + d];
    fb[3*H2 + d] = sm;
  }
}

__global__ void final_fc_kernel(const float* __restrict__ feat, const float* __restrict__ W,
                                const float* __restrict__ bias, float* __restrict__ out){
  int t = blockIdx.x*blockDim.x + threadIdx.x;
  if (t >= BB*POLN) return;
  int b = t / POLN, p = t % POLN;
  const float* f = feat + (long)b*(4*H2);
  const float* w = W + (long)p*(4*H2);
  float s = bias[p];
  for (int k=0;k<4*H2;k++) s += f[k]*w[k];
  out[t] = s;
}

extern "C" void kernel_launch(void* const* d_in, const int* in_sizes, int n_in,
                              void* d_out, int out_size, void* d_ws, size_t ws_size,
                              hipStream_t stream){
  (void)in_sizes; (void)n_in; (void)out_size; (void)ws_size;
  const int*   text_indices = (const int*)d_in[0];
  const int*   tran_starts  = (const int*)d_in[1];
  const int*   tran_ends    = (const int*)d_in[2];
  const int*   tran_len     = (const int*)d_in[3];
  const int*   span_starts  = (const int*)d_in[4];
  const int*   span_ends    = (const int*)d_in[5];
  const float* adj1 = (const float*)d_in[6];
  const float* adj2 = (const float*)d_in[7];
  const float* emb  = (const float*)d_in[8];
  const float* l2fW = (const float*)d_in[9];
  const float* l2fU = (const float*)d_in[10];
  const float* l2fb = (const float*)d_in[11];
  const float* l2bW = (const float*)d_in[12];
  const float* l2bU = (const float*)d_in[13];
  const float* l2bb = (const float*)d_in[14];
  const float* lfW  = (const float*)d_in[15];
  const float* lfU  = (const float*)d_in[16];
  const float* lfb  = (const float*)d_in[17];
  const float* lbW  = (const float*)d_in[18];
  const float* lbU  = (const float*)d_in[19];
  const float* lbb  = (const float*)d_in[20];
  const float* W_gc = (const float*)d_in[21];
  const float* b_gc = (const float*)d_in[22];
  const float* ln_g = (const float*)d_in[23];
  const float* ln_b = (const float*)d_in[24];
  const float* W_a  = (const float*)d_in[25];
  const float* b_a  = (const float*)d_in[26];
  const float* W_fc = (const float*)d_in[27];
  const float* b_fc = (const float*)d_in[28];
  float* out = (float*)d_out;

  char* base = (char*)d_ws; size_t off = 0;
  auto alloc = [&](size_t bytes) -> void* {
    off = (off + 255) & ~(size_t)255;
    void* p = base + off; off += bytes; return p;
  };

  // f16 weights
  f16* wl2fW = (f16*)alloc((size_t)H4*DD*2);
  f16* wl2fU = (f16*)alloc((size_t)H4*HH*2);
  f16* wl2bW = (f16*)alloc((size_t)H4*DD*2);
  f16* wl2bU = (f16*)alloc((size_t)H4*HH*2);
  f16* wlfW  = (f16*)alloc((size_t)H4*H2*2);
  f16* wlfU  = (f16*)alloc((size_t)H4*HH*2);
  f16* wlbW  = (f16*)alloc((size_t)H4*H2*2);
  f16* wlbU  = (f16*)alloc((size_t)H4*HH*2);
  f16* wWgc  = (f16*)alloc((size_t)H2*H2*2);
  f16* wWa   = (f16*)alloc((size_t)H2*H2*2);
  // activations / scratch
  f16*   x0b   = (f16*)  alloc((size_t)BB*SS*DD*2);
  int*   tlen  = (int*)  alloc((size_t)BB*4);
  float* xgf   = (float*)alloc((size_t)BB*SS*H4*4);
  float* xgb   = (float*)alloc((size_t)BB*SS*H4*4);
  float* gbuf  = (float*)alloc((size_t)BB*H4*4);
  float* hf1   = (float*)alloc((size_t)BB*HH*4);
  float* cf1   = (float*)alloc((size_t)BB*HH*4);
  float* hb1   = (float*)alloc((size_t)BB*HH*4);
  float* cb1   = (float*)alloc((size_t)BB*HH*4);
  float* hf2   = (float*)alloc((size_t)BB*HH*4);
  float* cf2   = (float*)alloc((size_t)BB*HH*4);
  float* hb2   = (float*)alloc((size_t)BB*HH*4);
  float* cb2   = (float*)alloc((size_t)BB*HH*4);
  f16*   hf1b  = (f16*)  alloc((size_t)BB*HH*2);
  f16*   hb1b  = (f16*)  alloc((size_t)BB*HH*2);
  f16*   hf2b  = (f16*)  alloc((size_t)BB*HH*2);
  f16*   hb2b  = (f16*)  alloc((size_t)BB*HH*2);
  float* out1c = (float*)alloc((size_t)BB*SS*H2*4);
  float* csbuf = (float*)alloc((size_t)BB*SS*H2*4);
  float* tmpsf = (float*)alloc((size_t)BB*NN*H2*4);
  f16*   tmpsh = (f16*)  alloc((size_t)BB*NN*H2*2);
  float* xg2f  = (float*)alloc((size_t)BB*NN*H4*4);
  float* xg2b  = (float*)alloc((size_t)BB*NN*H4*4);
  float* xseq  = (float*)alloc((size_t)BB*NN*H2*4);
  float* adjf  = (float*)alloc((size_t)BB*NN*NN*4);
  f16*   adjh  = (f16*)  alloc((size_t)BB*NN*NN*2);
  float* den   = (float*)alloc((size_t)BB*NN*4);
  float* gout  = (float*)alloc((size_t)BB*NN*H2*4);
  // goutb padded: logits GEMM B-operand may touch rows [NN, LGLD) of last batch
  f16*   goutb = (f16*)  alloc(((size_t)BB*NN + (LGLD-NN))*H2*2);
  f16*   goutT = (f16*)  alloc((size_t)BB*NN*H2*2);   // per-batch H2 x NN
  float* te    = (float*)alloc((size_t)BB*NN*H2*4);
  f16*   teT   = (f16*)  alloc((size_t)BB*NN*H2*2);   // per-batch H2 x NN
  float* tmp2  = (float*)alloc((size_t)BB*NN*H2*4);
  float* qbuf  = (float*)alloc((size_t)BB*NN*H2*4);
  f16*   qh    = (f16*)  alloc((size_t)BB*NN*H2*2);
  float* lgt   = (float*)alloc((size_t)BB*NN*LGLD*4); // padded ld
  f16*   atth  = (f16*)  alloc((size_t)BB*NN*NN*2);
  float* algn  = (float*)alloc((size_t)BB*NN*H2*4);
  float* textv = (float*)alloc((size_t)BB*NN*2*H2*4);
  float* feat  = (float*)alloc((size_t)BB*4*H2*4);

  auto CONV = [&](const float* s, f16* d, long n){
    f32_to_f16_kernel<<<dim3((unsigned)((n+255)/256)),256,0,stream>>>(s,d,n);
  };

  // weight conversions
  CONV(l2fW, wl2fW, (long)H4*DD);  CONV(l2fU, wl2fU, (long)H4*HH);
  CONV(l2bW, wl2bW, (long)H4*DD);  CONV(l2bU, wl2bU, (long)H4*HH);
  CONV(lfW,  wlfW,  (long)H4*H2);  CONV(lfU,  wlfU,  (long)H4*HH);
  CONV(lbW,  wlbW,  (long)H4*H2);  CONV(lbU,  wlbU,  (long)H4*HH);
  CONV(W_gc, wWgc,  (long)H2*H2);  CONV(W_a,  wWa,   (long)H2*H2);

  embed_kernel<<<BB*SS,256,0,stream>>>(text_indices, emb, x0b);
  textlen_kernel<<<BB,256,0,stream>>>(text_indices, tlen);

  // token-level BiLSTM input projections: M=8192, N=1536, K=768
  gemm_a_bt_t<8><<<dim3(H4/128,(BB*SS)/32,1),64,0,stream>>>(x0b, wl2fW, l2fb, xgf, BB*SS, H4, DD, 0,0,0);
  gemm_a_bt_t<8><<<dim3(H4/128,(BB*SS)/32,1),64,0,stream>>>(x0b, wl2bW, l2bb, xgb, BB*SS, H4, DD, 0,0,0);

  hipMemsetAsync(hf1,0,(size_t)BB*HH*4,stream); hipMemsetAsync(cf1,0,(size_t)BB*HH*4,stream);
  hipMemsetAsync(hb1,0,(size_t)BB*HH*4,stream); hipMemsetAsync(cb1,0,(size_t)BB*HH*4,stream);
  hipMemsetAsync(hf1b,0,(size_t)BB*HH*2,stream); hipMemsetAsync(hb1b,0,(size_t)BB*HH*2,stream);

  int upblocks = (BB*HH + 255)/256;
  for (int s=0;s<SS;s++){
    lstm_gate_gemm<<<H4/64,64,0,stream>>>(hf1b, wl2fU, xgf, tlen, gbuf, s, 0, SS);
    lstm_update<<<upblocks,256,0,stream>>>(gbuf, hf1, cf1, hf1b, out1c, tlen, s, 0, SS, H2, 0);
  }
  for (int s=0;s<SS;s++){
    lstm_gate_gemm<<<H4/64,64,0,stream>>>(hb1b, wl2bU, xgb, tlen, gbuf, s, 1, SS);
    lstm_update<<<upblocks,256,0,stream>>>(gbuf, hb1, cb1, hb1b, out1c, tlen, s, 1, SS, H2, HH);
  }

  cumsum_kernel<<<(BB*H2+255)/256,256,0,stream>>>(out1c, csbuf);
  tmps_kernel<<<(unsigned)(((long)BB*NN*H2+255)/256),256,0,stream>>>(csbuf, tran_starts, tran_ends, tran_len, tmpsf, tmpsh);

  // transition-level BiLSTM input projections: M=5120, N=1536, K=768
  gemm_a_bt_t<8><<<dim3(H4/128,(BB*NN)/32,1),64,0,stream>>>(tmpsh, wlfW, lfb, xg2f, BB*NN, H4, H2, 0,0,0);
  gemm_a_bt_t<8><<<dim3(H4/128,(BB*NN)/32,1),64,0,stream>>>(tmpsh, wlbW, lbb, xg2b, BB*NN, H4, H2, 0,0,0);

  hipMemsetAsync(hf2,0,(size_t)BB*HH*4,stream); hipMemsetAsync(cf2,0,(size_t)BB*HH*4,stream);
  hipMemsetAsync(hb2,0,(size_t)BB*HH*4,stream); hipMemsetAsync(cb2,0,(size_t)BB*HH*4,stream);
  hipMemsetAsync(hf2b,0,(size_t)BB*HH*2,stream); hipMemsetAsync(hb2b,0,(size_t)BB*HH*2,stream);

  for (int s=0;s<NN;s++){
    lstm_gate_gemm<<<H4/64,64,0,stream>>>(hf2b, wlfU, xg2f, tran_len, gbuf, s, 0, NN);
    lstm_update<<<upblocks,256,0,stream>>>(gbuf, hf2, cf2, hf2b, xseq, tran_len, s, 0, NN, H2, 0);
  }
  for (int s=0;s<NN;s++){
    lstm_gate_gemm<<<H4/64,64,0,stream>>>(hb2b, wlbU, xg2b, tran_len, gbuf, s, 1, NN);
    lstm_update<<<upblocks,256,0,stream>>>(gbuf, hb2, cb2, hb2b, xseq, tran_len, s, 1, NN, H2, HH);
  }

  adj_kernel<<<(unsigned)(((long)BB*NN*NN+255)/256),256,0,stream>>>(adj1, adj2, adjf, adjh);
  denom_kernel<<<BB*NN,256,0,stream>>>(adjf, den);

  hipMemcpyAsync(gout, xseq, (size_t)BB*NN*H2*4, hipMemcpyDeviceToDevice, stream);

  // 3 GCN layers
  for (int it=0; it<3; it++){
    CONV(gout, goutb, (long)BB*NN*H2);
    gemm_a_bt_t<8><<<dim3(H2/128,(BB*NN)/32,1),64,0,stream>>>(goutb, wWgc, nullptr, te, BB*NN, H2, H2, 0,0,0);
    // te[b]: NN x H2 (K-major for adj@te) -> transpose to H2 x NN so B loads are contiguous
    transpose_f32_f16<<<dim3(H2/32,NN/32,BB),256,0,stream>>>(te, teT, NN, H2);
    gemm_a_bt_t<8><<<dim3(H2/128,NN/32,BB),64,0,stream>>>(adjh, teT, nullptr, tmp2, NN, H2, NN,
                                                          (long)NN*NN, (long)NN*H2, (long)NN*H2);
    gcn_post_kernel<<<BB*NN,256,0,stream>>>(tmp2, den, b_gc, ln_g, ln_b, gout);
  }

  // attention
  CONV(gout, goutb, (long)BB*NN*H2);
  gemm_a_bt_t<8><<<dim3(H2/128,(BB*NN)/32,1),64,0,stream>>>(goutb, wWa, nullptr, qbuf, BB*NN, H2, H2, 0,0,0);
  CONV(qbuf, qh, (long)BB*NN*H2);
  // logits: N padded to LGLD=192 (cols >= NN are garbage, never read)
  gemm_a_bt_t<4><<<dim3(LGLD/64,NN/32,BB),64,0,stream>>>(qh, goutb, nullptr, lgt, NN, LGLD, H2,
                                                         (long)NN*H2, (long)NN*H2, (long)NN*LGLD);
  softmax_kernel<<<BB*NN,256,0,stream>>>(lgt, tran_len, atth);
  // att @ output: transpose gout -> H2 x NN, then fast-path GEMM
  transpose_f32_f16<<<dim3(H2/32,NN/32,BB),256,0,stream>>>(gout, goutT, NN, H2);
  gemm_a_bt_t<8><<<dim3(H2/128,NN/32,BB),64,0,stream>>>(atth, goutT, nullptr, algn, NN, H2, NN,
                                                        (long)NN*NN, (long)NN*H2, (long)NN*H2);
  text_kernel<<<(unsigned)(((long)BB*NN*H2+255)/256),256,0,stream>>>(algn, b_a, gout, tran_len, textv);

  span_feat_kernel<<<BB,256,0,stream>>>(textv, xseq, hf2, hb2, span_starts, span_ends, feat);
  final_fc_kernel<<<1,128,0,stream>>>(feat, W_fc, b_fc, out);
}
